// BERT_Network_87875030876798
// MI455X (gfx1250) — matrix operand order repaired and checked by reading, not compiled, not run on hardware
//
#include <hip/hip_runtime.h>
#include <hip/hip_bf16.h>

// ---- problem constants (match reference) ----
#define L_  6
#define H_  12
#define D_  768
#define F_  3072
#define V_  32000
#define S_  512
#define B_  4
#define DH_ 64
#define EPS_ 1e-6f

// ---- WMMA vector types (gfx1250, wave32) ----
typedef __attribute__((ext_vector_type(16))) __bf16 v16bf;
typedef __attribute__((ext_vector_type(2)))  __bf16 v2bf;
typedef __attribute__((ext_vector_type(8)))  float  v8f;

// Single f32 -> bf16 bits. Use a compiler-visible fptrunc so the backend can
// select the native gfx1250 convert (true16 v_cvt_bf16_f32) instead of a
// hand-rolled integer RNE sequence.
__device__ __forceinline__ unsigned short f2bf(float x) {
  union { __bf16 b; unsigned short u; } c;
  c.b = (__bf16)x;
  return c.u;
}

// Pack two f32 -> two bf16 in one dword. Prefer the packed-convert builtin if
// the toolchain declares it, else a v2bf fptrunc (backend can still fold both
// converts into one v_cvt_pk_bf16_f32).
__device__ __forceinline__ unsigned int pack2bf(float x, float y) {
#if __has_builtin(__builtin_amdgcn_cvt_pk_bf16_f32)
  union { v2bf v; unsigned int u; } c;
  c.v = __builtin_amdgcn_cvt_pk_bf16_f32(x, y);
  return c.u;
#else
  union { v2bf v; unsigned int u; } c;
  c.v.x = (__bf16)x;
  c.v.y = (__bf16)y;
  return c.u;
#endif
}

// ---- GEMM tiling ----
#define BM 128
#define BN 128
#define BK 32
#define LDSW 40   // padded bf16 row stride: 80 bytes, keeps every b128 frag load 16B aligned

// C[M,N] (f32) = alpha * A[M,K](f32,row) x B (f32) + bias, optional relu.
// BT=false: B is K x N row-major (ldb = row stride over K).
// BT=true : B is N x K row-major (ldb = row stride over N)  => computes A.B^T.
// Batched over blockIdx.z with (b,h) = (z/Hz, z%Hz) strides for attention heads.
template <bool BT>
__global__ __launch_bounds__(256) void gemm_bf16_wmma(
    const float* __restrict__ A, const float* __restrict__ Bm,
    float* __restrict__ C, const float* __restrict__ bias,
    int M, int N, int K, int lda, int ldb, int ldc,
    int Hz, long long sAb, long long sAh, long long sBb, long long sBh,
    long long sCb, long long sCh, float alpha, int relu)
{
  __shared__ alignas(16) unsigned short As[2][BM * LDSW];
  __shared__ alignas(16) unsigned short Bs[2][BN * LDSW];   // stored as [n][k]

  const int z  = blockIdx.z;
  const int bz = z / Hz, hz = z % Hz;
  A  += bz * sAb + hz * sAh;
  Bm += bz * sBb + hz * sBh;
  C  += bz * sCb + hz * sCh;

  const int m0   = blockIdx.y * BM;
  const int n0   = blockIdx.x * BN;
  const int tid  = threadIdx.x;
  const int lane = tid & 31;
  const int wave = tid >> 5;
  const int waveM = wave & 3;    // 4 row-groups of 32
  const int waveN = wave >> 2;   // 2 col-groups of 64
  const int half  = lane >> 4;
  const int l16   = lane & 15;

  v8f acc[2][4];
#pragma unroll
  for (int i = 0; i < 2; ++i)
#pragma unroll
    for (int j = 0; j < 4; ++j) { v8f zz = {}; acc[i][j] = zz; }

  auto stage = [&](int kt, int buf) {
    const int k0 = kt * BK;
    // ---- A tile: BM x BK, f32 -> bf16, row-major in LDS ----
#pragma unroll
    for (int it = 0; it < 4; ++it) {
      int f   = tid + it * 256;          // float4 index, 1024 total
      int row = f >> 3;                  // 8 float4 per row
      int col = (f & 7) << 2;
      float4 val = make_float4(0.f, 0.f, 0.f, 0.f);
      if (m0 + row < M) {
        const float* src = A + (long long)(m0 + row) * lda + (k0 + col);
        val = *reinterpret_cast<const float4*>(src);
        if (k0 + BK < K) __builtin_prefetch(src + BK, 0, 1);
      }
      unsigned short* dst = &As[buf][row * LDSW + col];
      *reinterpret_cast<unsigned int*>(dst)     = pack2bf(val.x, val.y);
      *reinterpret_cast<unsigned int*>(dst + 2) = pack2bf(val.z, val.w);
    }
    // ---- B tile -> Bs[n][k] ----
    if constexpr (BT) {
#pragma unroll
      for (int it = 0; it < 4; ++it) {
        int f   = tid + it * 256;
        int n   = f >> 3;
        int col = (f & 7) << 2;
        float4 val = make_float4(0.f, 0.f, 0.f, 0.f);
        if (n0 + n < N)
          val = *reinterpret_cast<const float4*>(Bm + (long long)(n0 + n) * ldb + (k0 + col));
        unsigned short* dst = &Bs[buf][n * LDSW + col];
        *reinterpret_cast<unsigned int*>(dst)     = pack2bf(val.x, val.y);
        *reinterpret_cast<unsigned int*>(dst + 2) = pack2bf(val.z, val.w);
      }
    } else {
#pragma unroll
      for (int it = 0; it < 4; ++it) {
        int f  = tid + it * 256;
        int kr = f >> 5;                 // 32 float4 per k-row (128 floats)
        int n  = (f & 31) << 2;
        float4 val = make_float4(0.f, 0.f, 0.f, 0.f);
        const float* src = Bm + (long long)(k0 + kr) * ldb;
        if (n0 + n + 3 < N) {
          val = *reinterpret_cast<const float4*>(src + n0 + n);
        } else {
          if (n0 + n     < N) val.x = src[n0 + n];
          if (n0 + n + 1 < N) val.y = src[n0 + n + 1];
          if (n0 + n + 2 < N) val.z = src[n0 + n + 2];
        }
        Bs[buf][(n + 0) * LDSW + kr] = f2bf(val.x);
        Bs[buf][(n + 1) * LDSW + kr] = f2bf(val.y);
        Bs[buf][(n + 2) * LDSW + kr] = f2bf(val.z);
        Bs[buf][(n + 3) * LDSW + kr] = f2bf(val.w);
      }
    }
  };

  auto compute = [&](int buf) {
    v16bf afr[2], bfr[4];
    // A frag (16x32 bf16): lane l16 = row, halves hold K 0..7/8..15 then 16..23/24..31
#pragma unroll
    for (int mi = 0; mi < 2; ++mi) {
      const unsigned short* base =
          &As[buf][(waveM * 32 + mi * 16 + l16) * LDSW + half * 8];
      uint4* d = reinterpret_cast<uint4*>(&afr[mi]);
      d[0] = *reinterpret_cast<const uint4*>(base);
      d[1] = *reinterpret_cast<const uint4*>(base + 16);
    }
    // B frag (32x16 bf16): lane l16 = col n, halves hold K 0..7/8..15 then 16..23/24..31
#pragma unroll
    for (int ni = 0; ni < 4; ++ni) {
      const unsigned short* base =
          &Bs[buf][(waveN * 64 + ni * 16 + l16) * LDSW + half * 8];
      uint4* d = reinterpret_cast<uint4*>(&bfr[ni]);
      d[0] = *reinterpret_cast<const uint4*>(base);
      d[1] = *reinterpret_cast<const uint4*>(base + 16);
    }
#pragma unroll
    for (int mi = 0; mi < 2; ++mi)
#pragma unroll
      for (int ni = 0; ni < 4; ++ni)
        acc[mi][ni] = __builtin_amdgcn_wmma_f32_16x16x32_bf16(
            false, afr[mi], false, bfr[ni], (short)0, acc[mi][ni], false, false);
  };

  const int nk = K / BK;
  stage(0, 0);
  int buf = 0;
  for (int kt = 0; kt < nk; ++kt) {
    __syncthreads();
    if (kt + 1 < nk) stage(kt + 1, buf ^ 1);
    compute(buf);
    buf ^= 1;
  }

  // ---- epilogue: C layout is VGPR r -> M = half*8 + r, N = l16 ----
#pragma unroll
  for (int mi = 0; mi < 2; ++mi) {
#pragma unroll
    for (int ni = 0; ni < 4; ++ni) {
      int col = n0 + waveN * 64 + ni * 16 + l16;
      if (col >= N) continue;
      float bv = bias ? bias[col] : 0.f;
#pragma unroll
      for (int r = 0; r < 8; ++r) {
        int row = m0 + waveM * 32 + mi * 16 + half * 8 + r;
        if (row >= M) continue;
        float vv = acc[mi][ni][r] * alpha + bv;
        if (relu) vv = vv > 0.f ? vv : 0.f;
        C[(long long)row * ldc + col] = vv;
      }
    }
  }
}

// ---- embedding gather: one block per (b,s) row ----
__global__ void embed_gather(const int* __restrict__ x, const float* __restrict__ W,
                             float* __restrict__ out) {
  int row = blockIdx.x;
  int idx = x[row];
  const float4* src = reinterpret_cast<const float4*>(W + (long long)idx * D_);
  float4* dst = reinterpret_cast<float4*>(out + (long long)row * D_);
  for (int c = threadIdx.x; c < D_ / 4; c += blockDim.x) dst[c] = src[c];
}

// ---- h = enc + pos (pos broadcast over batch) ----
__global__ void add_pos(const float* __restrict__ enc, const float* __restrict__ pos,
                        float* __restrict__ out, int n) {
  int i = blockIdx.x * blockDim.x + threadIdx.x;
  if (i < n) out[i] = enc[i] + pos[i % (S_ * D_)];
}

// ---- softmax over rows of length S_ (one block = one row, 256 threads x 2) ----
__global__ __launch_bounds__(256) void softmax_rows(float* __restrict__ x) {
  __shared__ float red[8], red2[8];
  float* row = x + (long long)blockIdx.x * S_;
  int t = threadIdx.x, lane = t & 31, wv = t >> 5;
  float a = row[t], b = row[t + 256];
  float mx = fmaxf(a, b);
  for (int off = 16; off > 0; off >>= 1) mx = fmaxf(mx, __shfl_xor(mx, off, 32));
  if (lane == 0) red[wv] = mx;
  __syncthreads();
  if (wv == 0) {
    float m2 = red[lane & 7];
    for (int off = 4; off > 0; off >>= 1) m2 = fmaxf(m2, __shfl_xor(m2, off, 32));
    if (lane == 0) red[0] = m2;
  }
  __syncthreads();
  mx = red[0];
  float ea = __expf(a - mx), eb = __expf(b - mx);
  float sm = ea + eb;
  for (int off = 16; off > 0; off >>= 1) sm += __shfl_xor(sm, off, 32);
  if (lane == 0) red2[wv] = sm;
  __syncthreads();
  if (wv == 0) {
    float s2 = red2[lane & 7];
    for (int off = 4; off > 0; off >>= 1) s2 += __shfl_xor(s2, off, 32);
    if (lane == 0) red2[0] = s2;
  }
  __syncthreads();
  float inv = 1.f / red2[0];
  row[t] = ea * inv;
  row[t + 256] = eb * inv;
}

// ---- out = res + LayerNorm(x)*scale + bias   (one block per row of D_=768) ----
__global__ __launch_bounds__(256) void ln_residual(
    const float* __restrict__ res, const float* __restrict__ x,
    const float* __restrict__ bias, const float* __restrict__ scale,
    float* __restrict__ out) {
  __shared__ float rs[8], rq[8];
  long long base = (long long)blockIdx.x * D_;
  int t = threadIdx.x, lane = t & 31, wv = t >> 5;
  float v0 = x[base + t], v1 = x[base + t + 256], v2 = x[base + t + 512];
  float s  = v0 + v1 + v2;
  float sq = v0 * v0 + v1 * v1 + v2 * v2;
  for (int off = 16; off > 0; off >>= 1) {
    s  += __shfl_xor(s, off, 32);
    sq += __shfl_xor(sq, off, 32);
  }
  if (lane == 0) { rs[wv] = s; rq[wv] = sq; }
  __syncthreads();
  if (wv == 0) {
    float aa = rs[lane & 7], bb = rq[lane & 7];
    for (int off = 4; off > 0; off >>= 1) {
      aa += __shfl_xor(aa, off, 32);
      bb += __shfl_xor(bb, off, 32);
    }
    if (lane == 0) { rs[0] = aa; rq[0] = bb; }
  }
  __syncthreads();
  float mean = rs[0] * (1.f / D_);
  float var  = rq[0] * (1.f / D_) - mean * mean;
  float rstd = rsqrtf(var + EPS_);
  out[base + t]       = res[base + t]       + (v0 - mean) * rstd * scale[t]       + bias[t];
  out[base + t + 256] = res[base + t + 256] + (v1 - mean) * rstd * scale[t + 256] + bias[t + 256];
  out[base + t + 512] = res[base + t + 512] + (v2 - mean) * rstd * scale[t + 512] + bias[t + 512];
}

extern "C" void kernel_launch(void* const* d_in, const int* in_sizes, int n_in,
                              void* d_out, int out_size, void* d_ws, size_t ws_size,
                              hipStream_t stream) {
  (void)in_sizes; (void)n_in; (void)out_size; (void)ws_size;
  const int*   x_input = (const int*)d_in[0];
  const float* W_emb   = (const float*)d_in[1];
  const float* p_q     = (const float*)d_in[2];
  const float* p_k     = (const float*)d_in[3];
  const float* p_v     = (const float*)d_in[4];
  const float* p_c     = (const float*)d_in[5];
  const float* p_ff1   = (const float*)d_in[6];
  const float* p_ff2   = (const float*)d_in[7];
  const float* b_ff1   = (const float*)d_in[8];
  const float* b_ff2   = (const float*)d_in[9];
  const float* lnb1    = (const float*)d_in[10];
  const float* lns1    = (const float*)d_in[11];
  const float* lnb2    = (const float*)d_in[12];
  const float* lns2    = (const float*)d_in[13];
  const float* o_bias  = (const float*)d_in[14];
  const float* o_scale = (const float*)d_in[15];
  const float* pos_emb = (const float*)d_in[16];

  const long long BSD = (long long)B_ * S_ * D_;   // 1,572,864
  const long long BSF = (long long)B_ * S_ * F_;   // 6,291,456
  float* ws   = (float*)d_ws;
  float* enc  = ws;
  float* h    = enc  + BSD;
  float* q    = h    + BSD;
  float* k    = q    + BSD;
  float* v    = k    + BSD;
  float* ctx  = v    + BSD;
  float* attn = ctx  + BSD;
  float* xs   = attn + BSD;
  float* f2   = xs   + BSD;
  float* xffw = f2   + BSD;
  float* f1   = xffw + BSD;
  float* sc   = f1   + BSF;                        // [B,H,S,S]

  float* logits = (float*)d_out;                   // [B,S,V]
  float* encout = logits + (long long)B_ * S_ * V_;// [B,S,D]

  const int MS = B_ * S_;                          // 2048
  const float inv_sqrt_dh = 0.125f;                // rsqrt(64)

  embed_gather<<<MS, 192, 0, stream>>>(x_input, W_emb, enc);

  const dim3 blk(256);
  const dim3 gDD(D_ / BN, MS / BM, 1);             // 6 x 16
  const dim3 gDF(F_ / BN, MS / BM, 1);             // 24 x 16
  const dim3 gSc(S_ / BN, S_ / BM, B_ * H_);       // 4 x 4 x 48
  const dim3 gCtx(1, S_ / BM, B_ * H_);            // 1 x 4 x 48
  const dim3 gLg(V_ / BN, MS / BM, 1);             // 250 x 16

  for (int m = 0; m < L_; ++m) {
    const float* Wq  = p_q   + (long long)m * D_ * D_;
    const float* Wk  = p_k   + (long long)m * D_ * D_;
    const float* Wv  = p_v   + (long long)m * D_ * D_;
    const float* Wc  = p_c   + (long long)m * D_ * D_;
    const float* Wf1 = p_ff1 + (long long)m * D_ * F_;
    const float* Wf2 = p_ff2 + (long long)m * F_ * D_;

    add_pos<<<((int)BSD + 255) / 256, blk, 0, stream>>>(
        enc, pos_emb + (long long)m * S_ * D_, h, (int)BSD);

    gemm_bf16_wmma<false><<<gDD, blk, 0, stream>>>(h, Wq, q, nullptr,
        MS, D_, D_, D_, D_, D_, 1, 0, 0, 0, 0, 0, 0, inv_sqrt_dh, 0);
    gemm_bf16_wmma<false><<<gDD, blk, 0, stream>>>(h, Wk, k, nullptr,
        MS, D_, D_, D_, D_, D_, 1, 0, 0, 0, 0, 0, 0, 1.f, 0);
    gemm_bf16_wmma<false><<<gDD, blk, 0, stream>>>(h, Wv, v, nullptr,
        MS, D_, D_, D_, D_, D_, 1, 0, 0, 0, 0, 0, 0, 1.f, 0);

    // scores[b,h] = q_head (SxDH) x k_head^T (DHxS)
    gemm_bf16_wmma<true><<<gSc, blk, 0, stream>>>(q, k, sc, nullptr,
        S_, S_, DH_, D_, D_, S_, H_,
        (long long)S_ * D_, DH_, (long long)S_ * D_, DH_,
        (long long)H_ * S_ * S_, (long long)S_ * S_, 1.f, 0);

    softmax_rows<<<B_ * H_ * S_, blk, 0, stream>>>(sc);

    // ctx[b,h] = alphas (SxS) x v_head (SxDH)
    gemm_bf16_wmma<false><<<gCtx, blk, 0, stream>>>(sc, v, ctx, nullptr,
        S_, DH_, S_, S_, D_, D_, H_,
        (long long)H_ * S_ * S_, (long long)S_ * S_,
        (long long)S_ * D_, DH_, (long long)S_ * D_, DH_, 1.f, 0);

    gemm_bf16_wmma<false><<<gDD, blk, 0, stream>>>(ctx, Wc, attn, nullptr,
        MS, D_, D_, D_, D_, D_, 1, 0, 0, 0, 0, 0, 0, 1.f, 0);

    ln_residual<<<MS, blk, 0, stream>>>(h, attn, lnb1 + (long long)m * D_,
                                        lns1 + (long long)m * D_, xs);

    gemm_bf16_wmma<false><<<gDF, blk, 0, stream>>>(xs, Wf1, f1,
        b_ff1 + (long long)m * F_,
        MS, F_, D_, D_, F_, F_, 1, 0, 0, 0, 0, 0, 0, 1.f, 1);

    gemm_bf16_wmma<false><<<gDD, blk, 0, stream>>>(f1, Wf2, f2,
        b_ff2 + (long long)m * D_,
        MS, D_, F_, F_, D_, D_, 1, 0, 0, 0, 0, 0, 0, 1.f, 0);

    ln_residual<<<MS, blk, 0, stream>>>(xs, f2, lnb2 + (long long)m * D_,
                                        lns2 + (long long)m * D_, xffw);
  }

  // enc_outputs = enc_inputs + LN(x_ffw_norm)
  ln_residual<<<MS, blk, 0, stream>>>(enc, xffw, o_bias, o_scale, encout);

  // logits = enc_outputs x W_emb^T   (NT, N = 32000)
  gemm_bf16_wmma<true><<<gLg, blk, 0, stream>>>(encout, W_emb, logits, nullptr,
      MS, V_, D_, D_, D_, V_, 1, 0, 0, 0, 0, 0, 0, 1.f, 0);
}
